// RHNNet_69939247448794
// MI455X (gfx1250) — matrix operand order repaired and checked by reading, not compiled, not run on hardware
//
#include <hip/hip_runtime.h>

// ---------------------------------------------------------------------------
// RHN (Recurrent Highway Network) for MI455X / gfx1250.
// B=64, T=256, IN=128, H=1024, OUT=128, DEPTH=2, 2 layers.
//
//  * All weights converted once/call to transposed bf16 (WT[n][k]); ~18.5 MB
//    -> resident in the 192 MB L2 for the entire recurrence.
//  * Every activation tensor is carried in BOTH fp32 (highway pass-through,
//    temporal state) and bf16 (WMMA A operand) so the hot K-loops do zero
//    fp32->bf16 conversion: A fragment = two contiguous global_load_b128.
//  * Fused micro-step: one 16x16 tile computes BOTH the transform-gate and
//    candidate GEMMs sharing the A fragment, then sigmoid/tanh/highway-mix
//    (+ temporal accumulate) in the epilogue.
//  * Serial-path latency: K split across 2 waves per tile (512 waves/step),
//    reduced through LDS. v_wmma_f32_16x16x32_bf16 everywhere.
// ---------------------------------------------------------------------------

typedef __bf16 bf16_t;
typedef __attribute__((ext_vector_type(16))) __bf16 v16bf;
typedef __attribute__((ext_vector_type(8)))  __bf16 v8bf;
typedef __attribute__((ext_vector_type(8)))  float  v8f;

static __device__ __forceinline__ v8f wmma_bf16(v16bf a, v16bf b, v8f c) {
  // 8 args: (neg_a, A, neg_b, B, c_mod, C, reuse_a, reuse_b)
  return __builtin_amdgcn_wmma_f32_16x16x32_bf16(false, a, false, b, (short)0, c,
                                                 false, false);
}

// A fragment (16x32): lanes 0-15 hold K = 0..15, lanes 16-31 hold K = 16..31
// (contiguous, matching the B fragment). One contiguous 32B load.
static __device__ __forceinline__ v16bf load_a_frag(const bf16_t* __restrict__ aRow,
                                                    int kk, int khalf) {
  const bf16_t* p = aRow + kk + khalf * 16;
  return *(const v16bf*)(p);
}

// ---------------------------------------------------------------------------
// fp32 W[K][N]  ->  bf16 WT[N][K]   (transpose + convert, weights only)
// ---------------------------------------------------------------------------
__global__ __launch_bounds__(256) void kxpose_bf16(const float* __restrict__ W,
                                                   bf16_t* __restrict__ WT,
                                                   int K, int N) {
  int total = K * N;
  for (int idx = blockIdx.x * blockDim.x + threadIdx.x; idx < total;
       idx += gridDim.x * blockDim.x) {
    int k = idx / N;
    int n = idx - k * N;
    WT[n * K + k] = (bf16_t)W[idx];
  }
}

// x [B,T,IN] fp32  ->  xb [T,B,IN] bf16  (time-major swizzle + convert)
__global__ __launch_bounds__(256) void kswz_x_bf16(const float* __restrict__ x,
                                                   bf16_t* __restrict__ xb,
                                                   int B, int T, int IN) {
  int total = B * T * IN;
  for (int idx = blockIdx.x * blockDim.x + threadIdx.x; idx < total;
       idx += gridDim.x * blockDim.x) {
    int r = idx / IN;       // r = t*B + b
    int i = idx - r * IN;
    int t = r / B;
    int b = r - t * B;
    xb[idx] = (bf16_t)x[(b * T + t) * IN + i];
  }
}

__global__ __launch_bounds__(256) void kzero_f32(float* __restrict__ p, int n) {
  for (int i = blockIdx.x * blockDim.x + threadIdx.x; i < n;
       i += gridDim.x * blockDim.x)
    p[i] = 0.0f;
}

// ---------------------------------------------------------------------------
// WMMA GEMM: C[r][n] = sum_k A[r][k]*WT[n][k] + bias[n];  Cb = bf16 shadow.
// A is row-major bf16. R,N multiples of 16; K multiple of 32.
// ---------------------------------------------------------------------------
__global__ __launch_bounds__(256) void kgemm_bias(
    const bf16_t* __restrict__ A, const bf16_t* __restrict__ WT,
    const float* __restrict__ bias, float* __restrict__ C,
    bf16_t* __restrict__ Cb, int R, int K, int N) {
  const int lane  = threadIdx.x & 31;
  const int ln    = lane & 15;
  const int khalf = lane >> 4;
  const int wavesTotal = (gridDim.x * blockDim.x) >> 5;
  const int wave0      = blockIdx.x * (blockDim.x >> 5) + (threadIdx.x >> 5);
  const int tilesM = R >> 4, tilesN = N >> 4;
  const int tiles  = tilesM * tilesN;

  for (int tile = wave0; tile < tiles; tile += wavesTotal) {
    const int tm  = tile / tilesN;
    const int tn  = tile - tm * tilesN;
    const int row = (tm << 4) + ln;
    const int col = (tn << 4) + ln;
    const bf16_t* aPtr = A + row * K;
    const bf16_t* bPtr = WT + col * K + khalf * 16;

    v8f acc = {};
    for (int kk = 0; kk < K; kk += 32) {
      __builtin_prefetch(bPtr + kk + 256, 0, 1);  // L2-resident weight stream
      v16bf afrag = load_a_frag(aPtr, kk, khalf);
      v16bf bfrag = *(const v16bf*)(bPtr + kk);   // 32B contiguous = B layout
      acc = wmma_bf16(afrag, bfrag, acc);
    }

    const float bv = bias[col];
#pragma unroll
    for (int r = 0; r < 8; ++r) {
      const int m   = (tm << 4) + r + (khalf << 3);  // C/D: M = r + 8*khalf
      const int off = m * N + col;
      const float v = acc[r] + bv;
      C[off] = v;
      if (Cb) Cb[off] = (bf16_t)v;
    }
  }
}

// ---------------------------------------------------------------------------
// One highway-depth micro-step (B=64, H=1024 fixed):
//   t   = sigmoid(h @ Wt + bt);  hl = tanh(h @ Wh + bh)
//   res = t*hl + (1-t)*h
// depth 0 (h_prev==nullptr): h_out=res, h_out_bf=(bf16)res
// depth 1 (h_prev!=nullptr): hn = res + h_prev; h_prev<-hn;
//                            h_out_bf=(bf16)hn; optionally out_bf=(bf16)hn
// 2 waves per 16x16 tile (K split in half), LDS partial reduction.
// Grid: 128 blocks x 128 threads = 512 waves = 256 tiles x 2 K-splits.
// ---------------------------------------------------------------------------
__global__ __launch_bounds__(128) void krhn_step(
    const bf16_t* __restrict__ a_bf, const float* __restrict__ h_in,
    const bf16_t* __restrict__ WtT, const float* __restrict__ bt,
    const bf16_t* __restrict__ WhT, const float* __restrict__ bh,
    float* __restrict__ h_out, bf16_t* __restrict__ h_out_bf,
    float* __restrict__ h_prev, bf16_t* __restrict__ out_bf) {
  constexpr int H = 1024;
  __shared__ float redT[2][256];
  __shared__ float redH[2][256];

  const int lane      = threadIdx.x & 31;
  const int ln        = lane & 15;
  const int khalf     = lane >> 4;
  const int waveInBlk = threadIdx.x >> 5;   // 0..3
  const int tileInWG  = waveInBlk >> 1;     // 0..1
  const int ks        = waveInBlk & 1;      // K-split half
  const int tile      = blockIdx.x * 2 + tileInWG;   // 0..255
  const int tm  = tile >> 6;                // 4 M-tiles
  const int tn  = tile & 63;                // 64 N-tiles
  const int row = (tm << 4) + ln;
  const int col = (tn << 4) + ln;

  const bf16_t* aPtr = a_bf + row * H;
  const bf16_t* wtP  = WtT + col * H + khalf * 16;
  const bf16_t* whP  = WhT + col * H + khalf * 16;

  const int k0 = ks * (H / 2), k1 = k0 + (H / 2);
  v8f tacc = {}, hacc = {};
  for (int kk = k0; kk < k1; kk += 32) {
    __builtin_prefetch(wtP + kk + 128, 0, 1);
    __builtin_prefetch(whP + kk + 128, 0, 1);
    v16bf afrag = load_a_frag(aPtr, kk, khalf);
    v16bf wtf = *(const v16bf*)(wtP + kk);
    v16bf whf = *(const v16bf*)(whP + kk);
    tacc = wmma_bf16(afrag, wtf, tacc);   // transform gate pre-activation
    hacc = wmma_bf16(afrag, whf, hacc);   // candidate pre-activation
  }

  if (ks == 1) {
#pragma unroll
    for (int r = 0; r < 8; ++r) {
      redT[tileInWG][lane * 8 + r] = tacc[r];
      redH[tileInWG][lane * 8 + r] = hacc[r];
    }
  }
  __syncthreads();
  if (ks == 1) return;

#pragma unroll
  for (int r = 0; r < 8; ++r) {
    tacc[r] += redT[tileInWG][lane * 8 + r];
    hacc[r] += redH[tileInWG][lane * 8 + r];
  }

  const float btv = bt[col];
  const float bhv = bh[col];
#pragma unroll
  for (int r = 0; r < 8; ++r) {
    const int m   = (tm << 4) + r + (khalf << 3);
    const int off = m * H + col;
    const float tg  = 1.0f / (1.0f + __expf(-(tacc[r] + btv)));
    const float hl  = tanhf(hacc[r] + bhv);
    const float hi  = h_in[off];
    const float res = tg * hl + (1.0f - tg) * hi;
    if (h_prev) {
      const float hn = res + h_prev[off];
      h_prev[off] = hn;
      h_out_bf[off] = (bf16_t)hn;           // bf16 shadow of temporal state
      if (out_bf) out_bf[off] = (bf16_t)hn; // layer-0 sequence output
    } else {
      h_out[off]    = res;
      h_out_bf[off] = (bf16_t)res;
    }
  }
}

// ---------------------------------------------------------------------------
// Host driver
// ---------------------------------------------------------------------------
extern "C" void kernel_launch(void* const* d_in, const int* in_sizes, int n_in,
                              void* d_out, int out_size, void* d_ws,
                              size_t ws_size, hipStream_t stream) {
  (void)in_sizes; (void)n_in; (void)out_size; (void)ws_size;
  constexpr int B = 64, T = 256, IN = 128, H = 1024, OUT = 128;

  const float* x      = (const float*)d_in[0];
  const float* l0_Win = (const float*)d_in[1];
  const float* l0_bin = (const float*)d_in[2];
  const float* l0_Wh  = (const float*)d_in[3];   // [2,H,H]
  const float* l0_bh  = (const float*)d_in[4];   // [2,H]
  const float* l0_Wt  = (const float*)d_in[5];
  const float* l0_bt  = (const float*)d_in[6];
  const float* l1_Win = (const float*)d_in[7];
  const float* l1_bin = (const float*)d_in[8];
  const float* l1_Wh  = (const float*)d_in[9];
  const float* l1_bh  = (const float*)d_in[10];
  const float* l1_Wt  = (const float*)d_in[11];
  const float* l1_bt  = (const float*)d_in[12];
  const float* fc_W   = (const float*)d_in[13];
  const float* fc_b   = (const float*)d_in[14];
  float* out = (float*)d_out;                    // [B,1,OUT] fp32

  // ---- workspace carve-up (~155 MB) ----
  char* ws = (char*)d_ws;
  size_t off = 0;
  auto take = [&](size_t bytes) -> char* {
    char* p = ws + off;
    off += (bytes + 255) & ~(size_t)255;
    return p;
  };
  bf16_t* wT_l0_in = (bf16_t*)take((size_t)IN * H * sizeof(bf16_t));
  bf16_t* wT_l1_in = (bf16_t*)take((size_t)H * H * sizeof(bf16_t));
  bf16_t *wT_l0_Wh[2], *wT_l0_Wt[2], *wT_l1_Wh[2], *wT_l1_Wt[2];
  for (int d = 0; d < 2; ++d) wT_l0_Wh[d] = (bf16_t*)take((size_t)H * H * sizeof(bf16_t));
  for (int d = 0; d < 2; ++d) wT_l0_Wt[d] = (bf16_t*)take((size_t)H * H * sizeof(bf16_t));
  for (int d = 0; d < 2; ++d) wT_l1_Wh[d] = (bf16_t*)take((size_t)H * H * sizeof(bf16_t));
  for (int d = 0; d < 2; ++d) wT_l1_Wt[d] = (bf16_t*)take((size_t)H * H * sizeof(bf16_t));
  bf16_t* wT_fc  = (bf16_t*)take((size_t)H * OUT * sizeof(bf16_t));
  bf16_t* xb     = (bf16_t*)take((size_t)T * B * IN * sizeof(bf16_t)); // [T,B,IN]
  float*  Z      = (float*)take((size_t)T * B * H * sizeof(float));    // [T,B,H]
  bf16_t* Zb     = (bf16_t*)take((size_t)T * B * H * sizeof(bf16_t));
  bf16_t* out0b  = (bf16_t*)take((size_t)T * B * H * sizeof(bf16_t));
  float*  hprev  = (float*)take((size_t)B * H * sizeof(float));
  bf16_t* hprevb = (bf16_t*)take((size_t)B * H * sizeof(bf16_t));
  float*  htmp   = (float*)take((size_t)B * H * sizeof(float));
  bf16_t* htmpb  = (bf16_t*)take((size_t)B * H * sizeof(bf16_t));

  // ---- weight convert+transpose (L2-resident thereafter) ----
  dim3 cb(256);
  kxpose_bf16<<<dim3(512), cb, 0, stream>>>(l0_Win, wT_l0_in, IN, H);
  kxpose_bf16<<<dim3(2048), cb, 0, stream>>>(l1_Win, wT_l1_in, H, H);
  for (int d = 0; d < 2; ++d) {
    kxpose_bf16<<<dim3(2048), cb, 0, stream>>>(l0_Wh + (size_t)d * H * H, wT_l0_Wh[d], H, H);
    kxpose_bf16<<<dim3(2048), cb, 0, stream>>>(l0_Wt + (size_t)d * H * H, wT_l0_Wt[d], H, H);
    kxpose_bf16<<<dim3(2048), cb, 0, stream>>>(l1_Wh + (size_t)d * H * H, wT_l1_Wh[d], H, H);
    kxpose_bf16<<<dim3(2048), cb, 0, stream>>>(l1_Wt + (size_t)d * H * H, wT_l1_Wt[d], H, H);
  }
  kxpose_bf16<<<dim3(512), cb, 0, stream>>>(fc_W, wT_fc, H, OUT);
  kswz_x_bf16<<<dim3(2048), cb, 0, stream>>>(x, xb, B, T, IN);

  // ---- layer 0: time-batched input projection Z0 = xb @ Win + bin ----
  kgemm_bias<<<dim3(2048), dim3(256), 0, stream>>>(xb, wT_l0_in, l0_bin, Z, Zb,
                                                   T * B, IN, H);

  // ---- layer 0 recurrence ----
  kzero_f32<<<dim3(256), cb, 0, stream>>>(hprev, B * H);
  for (int t = 0; t < T; ++t) {
    const size_t s = (size_t)t * B * H;
    krhn_step<<<dim3(128), dim3(128), 0, stream>>>(
        Zb + s, Z + s, wT_l0_Wt[0], l0_bt + 0 * H, wT_l0_Wh[0], l0_bh + 0 * H,
        htmp, htmpb, nullptr, nullptr);
    krhn_step<<<dim3(128), dim3(128), 0, stream>>>(
        htmpb, htmp, wT_l0_Wt[1], l0_bt + 1 * H, wT_l0_Wh[1], l0_bh + 1 * H,
        nullptr, hprevb, hprev, out0b + s);
  }

  // ---- layer 1: time-batched input projection Z1 = out0 @ Win + bin ----
  kgemm_bias<<<dim3(2048), dim3(256), 0, stream>>>(out0b, wT_l1_in, l1_bin, Z,
                                                   Zb, T * B, H, H);

  // ---- layer 1 recurrence (only final h needed) ----
  kzero_f32<<<dim3(256), cb, 0, stream>>>(hprev, B * H);
  for (int t = 0; t < T; ++t) {
    const size_t s = (size_t)t * B * H;
    krhn_step<<<dim3(128), dim3(128), 0, stream>>>(
        Zb + s, Z + s, wT_l1_Wt[0], l1_bt + 0 * H, wT_l1_Wh[0], l1_bh + 0 * H,
        htmp, htmpb, nullptr, nullptr);
    krhn_step<<<dim3(128), dim3(128), 0, stream>>>(
        htmpb, htmp, wT_l1_Wt[1], l1_bt + 1 * H, wT_l1_Wh[1], l1_bh + 1 * H,
        nullptr, hprevb, hprev, nullptr);
  }

  // ---- final FC on last hidden state: out = hprev @ fc_W + fc_b ----
  kgemm_bias<<<dim3(8), dim3(128), 0, stream>>>(hprevb, wT_fc, fc_b, out,
                                                (bf16_t*)nullptr, B, H, OUT);
}
